// HomogeneousGraphNeuralNetworkEnsemble_25211458028039
// MI455X (gfx1250) — matrix operand order repaired and checked by reading, not compiled, not run on hardware
//
#include <hip/hip_runtime.h>

typedef __attribute__((ext_vector_type(16))) __bf16 v16bf;
typedef __attribute__((ext_vector_type(8)))  __bf16 v8bf;
typedef __attribute__((ext_vector_type(8)))  float  v8f;

#define NENS   5
#define BATCH  1024
#define NOBJ   6
#define NNODES 7
#define SNODES (BATCH * NNODES)   // 7168
#define EPB    42
#define EEDGES (BATCH * EPB)      // 43008
#define EMB    128
#define HID    256
#define ACTD   8
#define KEP    288                // 264 padded to 32
#define KNP    416                // 392 padded to 32
#define LNEPS  1e-5f

// ---------------------------------------------------------------- WMMA helpers

__device__ __forceinline__ v8f wmma_bf16(v16bf a, v16bf b, v8f c) {
  // D = A(16x32 bf16) * B(32x16 bf16) + C(16x16 f32)
  return __builtin_amdgcn_wmma_f32_16x16x32_bf16(false, a, false, b, (short)0, c,
                                                 false, false);
}

// A-fragment (16x32 bf16) from row-major LDS tile.
// Layout: lanes 0-15 hold K=0..15 of row M=lane;
// lanes 16-31 hold K=16..31 of row M=lane-16 (matches load_b_sw).
// Lane data is one contiguous 32B run.
__device__ __forceinline__ v16bf load_a_lds(const __bf16* As, int ldk, int lane, int k0) {
  int m  = lane & 15;
  int kh = (lane >> 4) << 4;   // +0 or +16
  v16bf a;
#pragma unroll
  for (int v = 0; v < 8; ++v) {
    int k = k0 + kh + (v << 1);
    a[2 * v]     = As[m * ldk + k];
    a[2 * v + 1] = As[m * ldk + k + 1];
  }
  return a;
}

// B-fragment (32x16 bf16) from pre-swizzled weights.
// Swizzled layout: [Kp/32][Np][32]; element (k,c) at ((k>>5)*Np + c)*32 + (k&31).
// A lane's 16 values (fixed col, 16 consecutive K) are one contiguous 32B run.
__device__ __forceinline__ v16bf load_b_sw(const __bf16* Bsw, int Np, int lane,
                                           int ks, int nb) {
  const __bf16* p = Bsw + (((size_t)(ks >> 5) * Np + nb + (lane & 15)) << 5) +
                    ((lane >> 4) << 4);
  return *(const v16bf*)p;
}

// ---------------------------------------------------------------- weight prep

// f32 [K][N] -> bf16 swizzled [(Kp/32)][Np][32], zero padded.
__global__ __launch_bounds__(256)
void convert_swz_kernel(const float* __restrict__ src, __bf16* __restrict__ dst,
                        int K, int N, int Kp, int Np) {
  int e = blockIdx.y;
  int total = Kp * Np;
  for (int idx = blockIdx.x * blockDim.x + threadIdx.x; idx < total;
       idx += gridDim.x * blockDim.x) {
    int k = idx / Np, c = idx - k * Np;
    float v = (k < K && c < N) ? src[((size_t)e * K + k) * N + c] : 0.f;
    dst[(size_t)e * total + ((size_t)(k >> 5) * Np + c) * 32 + (k & 31)] = (__bf16)v;
  }
}

// ---------------------------------------------------------------- embeddings

__global__ __launch_bounds__(256)
void embed_agent_kernel(const float* __restrict__ agent, const __bf16* __restrict__ Wb,
                        const float* __restrict__ bias, __bf16* __restrict__ node_attr) {
  const int ens = blockIdx.y;
  const int mb  = blockIdx.x * 16;               // batch rows
  const int tid = threadIdx.x, lane = tid & 31, wv = tid >> 5;
  __shared__ __bf16 As[16 * 32];
  for (int idx = tid; idx < 16 * 32; idx += 256) {
    int r = idx >> 5, c = idx & 31;
    float v = (c < 16) ? agent[((size_t)ens * BATCH + (mb + r)) * 16 + c] : 0.f;
    As[idx] = (__bf16)v;
  }
  __syncthreads();
  const __bf16* Wb_e = Wb + (size_t)ens * 32 * EMB;
  int nb = wv * 16;                              // 8 waves * 16 cols = 128
  v16bf a = load_a_lds(As, 32, lane, 0);
  v16bf b = load_b_sw(Wb_e, EMB, lane, 0, nb);
  v8f c = {};
  c = wmma_bf16(a, b, c);
  int col = nb + (lane & 15);
  int mrow = (lane >> 4) << 3;
  const float* bias_e = bias + (size_t)ens * EMB;
#pragma unroll
  for (int v = 0; v < 8; ++v) {
    int bb = mb + mrow + v;
    node_attr[((size_t)ens * SNODES + (size_t)bb * NNODES) * EMB + col] =
        (__bf16)(c[v] + bias_e[col]);
  }
}

__global__ __launch_bounds__(256)
void embed_obj_kernel(const float* __restrict__ dyn, const float* __restrict__ stat,
                      const __bf16* __restrict__ Wb, const float* __restrict__ bias,
                      __bf16* __restrict__ node_attr) {
  const int ens = blockIdx.y;
  const int mb  = blockIdx.x * 16;               // object rows (B*NOBJ)
  const int tid = threadIdx.x, lane = tid & 31, wv = tid >> 5;
  __shared__ __bf16 As[16 * 32];
  for (int idx = tid; idx < 16 * 32; idx += 256) {
    int r = idx >> 5, c = idx & 31;
    int o = mb + r;
    int bb = o / NOBJ, k = o - bb * NOBJ;
    float v = 0.f;
    if (c < 12)      v = dyn[(((size_t)ens * BATCH + bb) * NOBJ + k) * 12 + c];
    else if (c < 16) v = stat[(((size_t)ens * BATCH + bb) * NOBJ + k) * 4 + (c - 12)];
    As[idx] = (__bf16)v;
  }
  __syncthreads();
  const __bf16* Wb_e = Wb + (size_t)ens * 32 * EMB;
  int nb = wv * 16;
  v16bf a = load_a_lds(As, 32, lane, 0);
  v16bf b = load_b_sw(Wb_e, EMB, lane, 0, nb);
  v8f c = {};
  c = wmma_bf16(a, b, c);
  int col = nb + (lane & 15);
  int mrow = (lane >> 4) << 3;
  const float* bias_e = bias + (size_t)ens * EMB;
#pragma unroll
  for (int v = 0; v < 8; ++v) {
    int o = mb + mrow + v;
    int bb = o / NOBJ, k = o - bb * NOBJ;
    node_attr[((size_t)ens * SNODES + (size_t)bb * NNODES + 1 + k) * EMB + col] =
        (__bf16)(c[v] + bias_e[col]);
  }
}

// -------------------------------------------------- fused edge MLP + aggregate
// Block = 48 edges (= 8 complete source-node groups): three 16-row WMMA tiles,
// segmented 6:1 reduction into LDS, single agg store (no HBM atomics).

__global__ __launch_bounds__(256)
void edge_mlp_kernel(const __bf16* __restrict__ node_attr, const float* __restrict__ action,
                     const __bf16* __restrict__ Weh, const float* __restrict__ beh,
                     const float* __restrict__ geh, const float* __restrict__ heh,
                     const __bf16* __restrict__ Weo, const float* __restrict__ beo,
                     __bf16* __restrict__ agg) {
  const int ens = blockIdx.y;
  const int tid = threadIdx.x, lane = tid & 31, wv = tid >> 5;
  const size_t nbase = (size_t)ens * SNODES;
  const __bf16* Weh_e = Weh + (size_t)ens * KEP * HID;
  const __bf16* Weo_e = Weo + (size_t)ens * HID * HID;
  const float* beh_e = beh + (size_t)ens * HID;
  const float* geh_e = geh + (size_t)ens * HID;
  const float* heh_e = heh + (size_t)ens * HID;
  const float* beo_e = beo + (size_t)ens * HID;

  __shared__ __bf16 As[16 * KEP];      // edge_in tile (bf16)
  __shared__ float  Hs[16 * HID];      // GEMM1 output (f32)
  __shared__ __bf16 A2[16 * HID];      // post-LN/ReLU (bf16)
  __shared__ float  red1[16 * 16], red2[16 * 16];
  __shared__ float  mu_s[16], rs_s[16];
  __shared__ float  aggs[8 * HID];     // 8 source nodes per block

  for (int idx = tid; idx < 8 * HID; idx += 256) aggs[idx] = 0.f;
  __syncthreads();

  const int nb = wv * 32;              // each wave: 32 output cols

  for (int t = 0; t < 3; ++t) {
    int gbase = blockIdx.x * 48 + t * 16;
    // gather edge_in = [x[src] | x[tgt] | action | pad] -> LDS, 8-elem vectors
    for (int u = tid; u < 16 * (KEP / 8); u += 256) {
      int r  = u / (KEP / 8);
      int cu = (u - r * (KEP / 8)) * 8;
      int g  = gbase + r;
      int bb = g / EPB, pl = g - bb * EPB;
      int i  = pl / 6,  jp = pl - i * 6;
      int j  = jp + (jp >= i ? 1 : 0);
      v8bf v = {};
      if (cu < EMB)
        v = *(const v8bf*)&node_attr[(nbase + (size_t)bb * NNODES + i) * EMB + cu];
      else if (cu < 2 * EMB)
        v = *(const v8bf*)&node_attr[(nbase + (size_t)bb * NNODES + j) * EMB + (cu - EMB)];
      else if (cu < 2 * EMB + ACTD) {
        const float* ap = &action[((size_t)ens * BATCH + bb) * ACTD];
#pragma unroll
        for (int q2 = 0; q2 < 8; ++q2) v[q2] = (__bf16)ap[q2];
      }
      *(v8bf*)&As[r * KEP + cu] = v;
    }
    __syncthreads();

    // GEMM1: (16 x 288) @ (288 x 256)
    v8f acc0 = {}, acc1 = {};
    for (int ks = 0; ks < KEP; ks += 32) {
      v16bf a  = load_a_lds(As, KEP, lane, ks);
      v16bf b0 = load_b_sw(Weh_e, HID, lane, ks, nb);
      acc0 = wmma_bf16(a, b0, acc0);
      v16bf b1 = load_b_sw(Weh_e, HID, lane, ks, nb + 16);
      acc1 = wmma_bf16(a, b1, acc1);
    }
    {
      int colA = nb + (lane & 15), colB = colA + 16;
      int mrow = (lane >> 4) << 3;
#pragma unroll
      for (int v = 0; v < 8; ++v) {
        Hs[(mrow + v) * HID + colA] = acc0[v] + beh_e[colA];
        Hs[(mrow + v) * HID + colB] = acc1[v] + beh_e[colB];
      }
    }
    __syncthreads();

    // LayerNorm + ReLU (16 threads per row)
    {
      int r = tid >> 4, lc = tid & 15;
      float s1 = 0.f, s2 = 0.f;
      for (int c2 = lc; c2 < HID; c2 += 16) {
        float x = Hs[r * HID + c2]; s1 += x; s2 += x * x;
      }
      red1[r * 16 + lc] = s1; red2[r * 16 + lc] = s2;
      __syncthreads();
      if (lc == 0) {
        float a = 0.f, b = 0.f;
        for (int k2 = 0; k2 < 16; ++k2) { a += red1[r * 16 + k2]; b += red2[r * 16 + k2]; }
        float mu = a * (1.f / HID);
        mu_s[r] = mu;
        rs_s[r] = rsqrtf(b * (1.f / HID) - mu * mu + LNEPS);
      }
      __syncthreads();
      float mu = mu_s[r], rs = rs_s[r];
      for (int c2 = lc; c2 < HID; c2 += 16) {
        float x = (Hs[r * HID + c2] - mu) * rs * geh_e[c2] + heh_e[c2];
        A2[r * HID + c2] = (__bf16)fmaxf(x, 0.f);
      }
    }
    __syncthreads();

    // GEMM2: (16 x 256) @ (256 x 256), then segmented-sum into aggs (6 edges/node)
    v8f o0 = {}, o1 = {};
    for (int ks = 0; ks < HID; ks += 32) {
      v16bf a  = load_a_lds(A2, HID, lane, ks);
      v16bf b0 = load_b_sw(Weo_e, HID, lane, ks, nb);
      o0 = wmma_bf16(a, b0, o0);
      v16bf b1 = load_b_sw(Weo_e, HID, lane, ks, nb + 16);
      o1 = wmma_bf16(a, b1, o1);
    }
    {
      int colA = nb + (lane & 15), colB = colA + 16;
      int mrow = (lane >> 4) << 3;
#pragma unroll
      for (int v = 0; v < 8; ++v) {
        int rr = t * 16 + mrow + v;
        int q  = rr / 6;                 // local source node (0..7)
        atomicAdd(&aggs[q * HID + colA], o0[v]);
        atomicAdd(&aggs[q * HID + colB], o1[v]);
      }
    }
    __syncthreads();
  }

  // agg = sums/6 + beo  (beo folded in once: mean of 6 identical biases)
  int q0 = blockIdx.x * 8;
  for (int idx = tid; idx < 8 * HID; idx += 256) {
    int q = idx / HID, cc = idx - q * HID;
    float v = aggs[idx] * (1.f / 6.f) + beo_e[cc];
    agg[(nbase + q0 + q) * HID + cc] = (__bf16)v;
  }
}

// ---------------------------------------------------------- fused node MLP

__global__ __launch_bounds__(256)
void node_mlp_kernel(const __bf16* __restrict__ node_attr, const float* __restrict__ action,
                     const __bf16* __restrict__ agg,
                     const __bf16* __restrict__ Wnh, const float* __restrict__ bnh,
                     const float* __restrict__ gnh, const float* __restrict__ hnh,
                     const __bf16* __restrict__ Wno, const float* __restrict__ bno,
                     __bf16* __restrict__ x2) {
  const int ens = blockIdx.y;
  const int tid = threadIdx.x, lane = tid & 31, wv = tid >> 5;
  const size_t nbase = (size_t)ens * SNODES;
  const __bf16* Wnh_e = Wnh + (size_t)ens * KNP * HID;
  const __bf16* Wno_e = Wno + (size_t)ens * HID * EMB;
  const float* bnh_e = bnh + (size_t)ens * HID;
  const float* gnh_e = gnh + (size_t)ens * HID;
  const float* hnh_e = hnh + (size_t)ens * HID;
  const float* bno_e = bno + (size_t)ens * EMB;

  __shared__ __bf16 As[16 * KNP];
  __shared__ float  Hs[16 * HID];
  __shared__ __bf16 A2[16 * HID];
  __shared__ float  red1[16 * 16], red2[16 * 16];
  __shared__ float  mu_s[16], rs_s[16];

  const int sbase = blockIdx.x * 16;
  // node_in = [x | action | agg | pad] -> LDS, 8-elem vectors
  for (int u = tid; u < 16 * (KNP / 8); u += 256) {
    int r  = u / (KNP / 8);
    int cu = (u - r * (KNP / 8)) * 8;
    int s  = sbase + r;
    int bb = s / NNODES;
    v8bf v = {};
    if (cu < EMB)
      v = *(const v8bf*)&node_attr[(nbase + s) * EMB + cu];
    else if (cu < EMB + ACTD) {
      const float* ap = &action[((size_t)ens * BATCH + bb) * ACTD];
#pragma unroll
      for (int q2 = 0; q2 < 8; ++q2) v[q2] = (__bf16)ap[q2];
    } else if (cu < EMB + ACTD + HID)
      v = *(const v8bf*)&agg[(nbase + s) * HID + (cu - EMB - ACTD)];
    *(v8bf*)&As[r * KNP + cu] = v;
  }
  __syncthreads();

  // GEMM1: (16 x 416) @ (416 x 256)
  const int nb = wv * 32;
  v8f acc0 = {}, acc1 = {};
  for (int ks = 0; ks < KNP; ks += 32) {
    v16bf a  = load_a_lds(As, KNP, lane, ks);
    v16bf b0 = load_b_sw(Wnh_e, HID, lane, ks, nb);
    acc0 = wmma_bf16(a, b0, acc0);
    v16bf b1 = load_b_sw(Wnh_e, HID, lane, ks, nb + 16);
    acc1 = wmma_bf16(a, b1, acc1);
  }
  {
    int colA = nb + (lane & 15), colB = colA + 16;
    int mrow = (lane >> 4) << 3;
#pragma unroll
    for (int v = 0; v < 8; ++v) {
      Hs[(mrow + v) * HID + colA] = acc0[v] + bnh_e[colA];
      Hs[(mrow + v) * HID + colB] = acc1[v] + bnh_e[colB];
    }
  }
  __syncthreads();

  // LayerNorm + ReLU
  {
    int r = tid >> 4, lc = tid & 15;
    float s1 = 0.f, s2 = 0.f;
    for (int c2 = lc; c2 < HID; c2 += 16) {
      float x = Hs[r * HID + c2]; s1 += x; s2 += x * x;
    }
    red1[r * 16 + lc] = s1; red2[r * 16 + lc] = s2;
    __syncthreads();
    if (lc == 0) {
      float a = 0.f, b = 0.f;
      for (int k2 = 0; k2 < 16; ++k2) { a += red1[r * 16 + k2]; b += red2[r * 16 + k2]; }
      float mu = a * (1.f / HID);
      mu_s[r] = mu;
      rs_s[r] = rsqrtf(b * (1.f / HID) - mu * mu + LNEPS);
    }
    __syncthreads();
    float mu = mu_s[r], rs = rs_s[r];
    for (int c2 = lc; c2 < HID; c2 += 16) {
      float x = (Hs[r * HID + c2] - mu) * rs * gnh_e[c2] + hnh_e[c2];
      A2[r * HID + c2] = (__bf16)fmaxf(x, 0.f);
    }
  }
  __syncthreads();

  // GEMM2: (16 x 256) @ (256 x 128), one 16-col tile per wave
  const int nb2 = wv * 16;
  v8f oacc = {};
  for (int ks = 0; ks < HID; ks += 32) {
    v16bf a = load_a_lds(A2, HID, lane, ks);
    v16bf b = load_b_sw(Wno_e, EMB, lane, ks, nb2);
    oacc = wmma_bf16(a, b, oacc);
  }
  {
    int col = nb2 + (lane & 15);
    int mrow = (lane >> 4) << 3;
#pragma unroll
    for (int v = 0; v < 8; ++v) {
      int s = sbase + mrow + v;
      x2[(nbase + s) * EMB + col] = (__bf16)(oacc[v] + bno_e[col]);
    }
  }
}

// ---------------------------------------------------------------- output heads

__global__ __launch_bounds__(256)
void out_agent_kernel(const __bf16* __restrict__ x2, const __bf16* __restrict__ Wb,
                      const float* __restrict__ bias, float* __restrict__ out) {
  const int ens = blockIdx.y;
  const int tid = threadIdx.x, lane = tid & 31, wv = tid >> 5;
  const int rbase = blockIdx.x * 128;
  const size_t nbase = (size_t)ens * SNODES;
  __shared__ __bf16 As[128 * EMB];
  for (int u = tid; u < 128 * (EMB / 8); u += 256) {
    int r = u >> 4, cu = (u & 15) * 8;
    int bb = rbase + r;
    *(v8bf*)&As[r * EMB + cu] =
        *(const v8bf*)&x2[(nbase + (size_t)bb * NNODES) * EMB + cu];
  }
  __syncthreads();
  const __bf16* Wb_e = Wb + (size_t)ens * EMB * 16;
  v8f acc = {};
  for (int ks = 0; ks < EMB; ks += 32) {
    v16bf a = load_a_lds(As + wv * 16 * EMB, EMB, lane, ks);
    v16bf b = load_b_sw(Wb_e, 16, lane, ks, 0);
    acc = wmma_bf16(a, b, acc);
  }
  int col = lane & 15;
  int mrow = (lane >> 4) << 3;
  const float* bias_e = bias + (size_t)ens * 16;
#pragma unroll
  for (int v = 0; v < 8; ++v) {
    int row = rbase + wv * 16 + mrow + v;
    out[((size_t)ens * BATCH + row) * 16 + col] = acc[v] + bias_e[col];
  }
}

__global__ __launch_bounds__(256)
void out_obj_kernel(const __bf16* __restrict__ x2, const __bf16* __restrict__ Wb,
                    const float* __restrict__ bias, float* __restrict__ out) {
  const int ens = blockIdx.y;
  const int tid = threadIdx.x, lane = tid & 31, wv = tid >> 5;
  const int rbase = blockIdx.x * 128;
  const size_t nbase = (size_t)ens * SNODES;
  __shared__ __bf16 As[128 * EMB];
  for (int u = tid; u < 128 * (EMB / 8); u += 256) {
    int r = u >> 4, cu = (u & 15) * 8;
    int o = rbase + r;
    int bb = o / NOBJ, k = o - bb * NOBJ;
    *(v8bf*)&As[r * EMB + cu] =
        *(const v8bf*)&x2[(nbase + (size_t)bb * NNODES + 1 + k) * EMB + cu];
  }
  __syncthreads();
  const __bf16* Wb_e = Wb + (size_t)ens * EMB * 16;   // padded to 16 cols
  v8f acc = {};
  for (int ks = 0; ks < EMB; ks += 32) {
    v16bf a = load_a_lds(As + wv * 16 * EMB, EMB, lane, ks);
    v16bf b = load_b_sw(Wb_e, 16, lane, ks, 0);
    acc = wmma_bf16(a, b, acc);
  }
  int col = lane & 15;
  int mrow = (lane >> 4) << 3;
  const float* bias_e = bias + (size_t)ens * 12;
  if (col < 12) {
#pragma unroll
    for (int v = 0; v < 8; ++v) {
      int row = rbase + wv * 16 + mrow + v;
      out[((size_t)ens * BATCH * NOBJ + row) * 12 + col] = acc[v] + bias_e[col];
    }
  }
}

// ---------------------------------------------------------------- launch

extern "C" void kernel_launch(void* const* d_in, const int* in_sizes, int n_in,
                              void* d_out, int out_size, void* d_ws, size_t ws_size,
                              hipStream_t stream) {
  const float* agent   = (const float*)d_in[0];
  const float* objdyn  = (const float*)d_in[1];
  const float* objstat = (const float*)d_in[2];
  const float* action  = (const float*)d_in[3];
  const float* We_ag = (const float*)d_in[4];  const float* be_ag = (const float*)d_in[5];
  const float* We_ob = (const float*)d_in[6];  const float* be_ob = (const float*)d_in[7];
  const float* Weh   = (const float*)d_in[8];  const float* beh   = (const float*)d_in[9];
  const float* geh   = (const float*)d_in[10]; const float* heh   = (const float*)d_in[11];
  const float* Weo   = (const float*)d_in[12]; const float* beo   = (const float*)d_in[13];
  const float* Wnh   = (const float*)d_in[14]; const float* bnh   = (const float*)d_in[15];
  const float* gnh   = (const float*)d_in[16]; const float* hnh   = (const float*)d_in[17];
  const float* Wno   = (const float*)d_in[18]; const float* bno   = (const float*)d_in[19];
  const float* Wo_ag = (const float*)d_in[20]; const float* bo_ag = (const float*)d_in[21];
  const float* Wo_ob = (const float*)d_in[22]; const float* bo_ob = (const float*)d_in[23];
  float* out = (float*)d_out;

  char* base = (char*)d_ws;
  size_t off = 0;
  auto take = [&](size_t elems) -> __bf16* {
    __bf16* p = (__bf16*)(base + off);
    off += (elems * sizeof(__bf16) + 255) & ~(size_t)255;
    return p;
  };
  __bf16* wag = take((size_t)NENS * 32 * EMB);
  __bf16* wob = take((size_t)NENS * 32 * EMB);
  __bf16* weh = take((size_t)NENS * KEP * HID);
  __bf16* weo = take((size_t)NENS * HID * HID);
  __bf16* wnh = take((size_t)NENS * KNP * HID);
  __bf16* wno = take((size_t)NENS * HID * EMB);
  __bf16* woa = take((size_t)NENS * EMB * 16);
  __bf16* woo = take((size_t)NENS * EMB * 16);
  __bf16* nattr = take((size_t)NENS * SNODES * EMB);
  __bf16* aggb  = take((size_t)NENS * SNODES * HID);
  __bf16* x2b   = take((size_t)NENS * SNODES * EMB);

  dim3 blk(256);
  convert_swz_kernel<<<dim3(32, NENS), blk, 0, stream>>>(We_ag, wag, 16, 128, 32, 128);
  convert_swz_kernel<<<dim3(32, NENS), blk, 0, stream>>>(We_ob, wob, 16, 128, 32, 128);
  convert_swz_kernel<<<dim3(32, NENS), blk, 0, stream>>>(Weh,   weh, 264, 256, KEP, 256);
  convert_swz_kernel<<<dim3(32, NENS), blk, 0, stream>>>(Weo,   weo, 256, 256, 256, 256);
  convert_swz_kernel<<<dim3(32, NENS), blk, 0, stream>>>(Wnh,   wnh, 392, 256, KNP, 256);
  convert_swz_kernel<<<dim3(32, NENS), blk, 0, stream>>>(Wno,   wno, 256, 128, 256, 128);
  convert_swz_kernel<<<dim3(32, NENS), blk, 0, stream>>>(Wo_ag, woa, 128, 16, 128, 16);
  convert_swz_kernel<<<dim3(32, NENS), blk, 0, stream>>>(Wo_ob, woo, 128, 12, 128, 16);

  embed_agent_kernel<<<dim3(BATCH / 16, NENS), blk, 0, stream>>>(agent, wag, be_ag, nattr);
  embed_obj_kernel<<<dim3((BATCH * NOBJ) / 16, NENS), blk, 0, stream>>>(objdyn, objstat,
                                                                        wob, be_ob, nattr);
  edge_mlp_kernel<<<dim3(EEDGES / 48, NENS), blk, 0, stream>>>(nattr, action, weh, beh,
                                                               geh, heh, weo, beo, aggb);
  node_mlp_kernel<<<dim3(SNODES / 16, NENS), blk, 0, stream>>>(nattr, action, aggb, wnh,
                                                               bnh, gnh, hnh, wno, bno, x2b);
  out_agent_kernel<<<dim3(BATCH / 128, NENS), blk, 0, stream>>>(x2b, woa, bo_ag, out);
  out_obj_kernel<<<dim3((BATCH * NOBJ) / 128, NENS), blk, 0, stream>>>(
      x2b, woo, bo_ob, out + (size_t)NENS * BATCH * 16);
}
